// PVToBEV_23459111371117
// MI455X (gfx1250) — compile-verified, matrix operand order checked by reading, not run on hardware
//
#include <hip/hip_runtime.h>
#include <hip/hip_bf16.h>
#include <stdint.h>

// ---------------------------------------------------------------------------
// Types
// ---------------------------------------------------------------------------
typedef __bf16 bf16;
typedef __attribute__((ext_vector_type(16))) __bf16 v16bf;
typedef __attribute__((ext_vector_type(4)))  __bf16 v4bf;
typedef __attribute__((ext_vector_type(8)))  float  v8f;

#define USE_ASYNC_LDS 1

__device__ __forceinline__ float gelu_exact(float x) {
    return 0.5f * x * (1.0f + erff(x * 0.70710678118654752f));
}
__device__ __forceinline__ float wave_sum32(float v) {
    #pragma unroll
    for (int o = 16; o > 0; o >>= 1) v += __shfl_xor(v, o, 32);
    return v;
}
__device__ __forceinline__ float wave_max32(float v) {
    #pragma unroll
    for (int o = 16; o > 0; o >>= 1) v = fmaxf(v, __shfl_xor(v, o, 32));
    return v;
}

// ---------------------------------------------------------------------------
// Model constants
// ---------------------------------------------------------------------------
#define CB   2      // batch
#define CN   6      // cameras
#define CC   128    // DIM
#define CHF  128    // feature H
#define CWF  352    // feature W
#define CHB  40     // BEV H
#define CWB  40     // BEV W
#define CZP  16     // Z points
#define CT   96     // tokens = N*ZP
#define CHW  1600   // BEV HW
#define CBHW 3200   // B*HW
#define CH4  512    // 4*DIM
#define CSR  100    // reduced seq (10*10)
#define CBSR 200    // B * reduced seq

// ---------------------------------------------------------------------------
// 1) fmaps NCHW f32 -> NHWC bf16 (channel-last), LDS tile transpose
// ---------------------------------------------------------------------------
__global__ void k_nchw_to_nhwc_bf16(const float* __restrict__ in,
                                    bf16* __restrict__ out,
                                    int C, int HW) {
    __shared__ float tile[32][33];
    int p0 = blockIdx.x * 32;
    int c0 = blockIdx.y * 32;
    int bn = blockIdx.z;
    int tx = threadIdx.x & 31, ty = threadIdx.x >> 5;   // ty: 0..7
    #pragma unroll
    for (int i = 0; i < 4; i++) {
        int c = c0 + ty + i * 8;
        int p = p0 + tx;
        float v = (p < HW) ? in[((size_t)bn * C + c) * HW + p] : 0.f;
        tile[ty + i * 8][tx] = v;                       // tile[c][p]
    }
    __syncthreads();
    #pragma unroll
    for (int i = 0; i < 4; i++) {
        int p = p0 + ty + i * 8;
        int c = c0 + tx;
        if (p < HW) out[((size_t)bn * HW + p) * C + c] = (bf16)tile[tx][ty + i * 8];
    }
}

// ---------------------------------------------------------------------------
// 2) mv tokens: project BEV columns into views, bilinear gather (channel-last
//    bf16 source => every tap is a contiguous 256B line). One wave per
//    (b, n, point); 4 channels per lane.
// ---------------------------------------------------------------------------
__global__ __launch_bounds__(256)
void k_mv_gather(const bf16* __restrict__ fm,      // (B*N, Hf, Wf, C) bf16
                 const float* __restrict__ extr,   // (B,N,4,4)
                 const float* __restrict__ intr,   // (B,N,3,3)
                 const float* __restrict__ res,    // (B,N,2)
                 const float* __restrict__ mpp,    // (1,)
                 bf16* __restrict__ mv)            // (B, HW, T, C) bf16
{
    const int P = CHB * CWB * CZP;                 // 25600
    int wid = blockIdx.x * 8 + (threadIdx.x >> 5);
    if (wid >= CB * CN * P) return;
    int lane = threadIdx.x & 31;

    int b = wid / (CN * P);
    int r = wid - b * (CN * P);
    int n = r / P;
    int p = r - n * P;
    int yi = p / (CWB * CZP);
    int r2 = p - yi * (CWB * CZP);
    int xi = r2 / CZP;
    int zi = r2 - xi * CZP;

    float cell = mpp[0] * 8.0f;                    // m_per_pixel * 2^UP
    float X = (xi - (CWB - 1) * 0.5f) * cell;
    float Y = (yi - (CHB - 1) * 0.5f) * cell;
    float Z = -5.0f + (float)zi * (15.0f / (CZP - 1));

    const float* e = extr + ((size_t)(b * CN + n)) * 16;
    float c0 = e[0] * X + e[1] * Y + e[2]  * Z + e[3];
    float c1 = e[4] * X + e[5] * Y + e[6]  * Z + e[7];
    float c2 = e[8] * X + e[9] * Y + e[10] * Z + e[11];
    const float* km = intr + ((size_t)(b * CN + n)) * 9;
    float u = km[0] * c0 + km[1] * c1 + km[2] * c2;
    float v = km[3] * c0 + km[4] * c1 + km[5] * c2;
    float w = km[6] * c0 + km[7] * c1 + km[8] * c2;
    bool valid = w > 1e-3f;
    float d = fmaxf(w, 1e-3f);
    const float* rr = res + ((size_t)(b * CN + n)) * 2;
    float xf = u / d / rr[0] * (float)CWF - 0.5f;
    float yf = v / d / rr[1] * (float)CHF - 0.5f;

    int x0 = (int)floorf(xf), y0 = (int)floorf(yf);
    float fx = xf - (float)x0, fy = yf - (float)y0;
    float w00 = (1.f - fy) * (1.f - fx), w01 = (1.f - fy) * fx;
    float w10 = fy * (1.f - fx),         w11 = fy * fx;

    int c4 = lane * 4;
    const bf16* base = fm + ((size_t)(b * CN + n) * CHF * CWF) * CC + c4;
    float a0 = 0.f, a1 = 0.f, a2 = 0.f, a3 = 0.f;
    #pragma unroll
    for (int t = 0; t < 4; t++) {
        int yy = y0 + (t >> 1);
        int xx = x0 + (t & 1);
        float wt = (t == 0) ? w00 : (t == 1) ? w01 : (t == 2) ? w10 : w11;
        bool inb = valid && xx >= 0 && xx < CWF && yy >= 0 && yy < CHF;
        if (inb) {
            v4bf vv = *(const v4bf*)(base + ((size_t)yy * CWF + xx) * CC);
            a0 += wt * (float)vv[0]; a1 += wt * (float)vv[1];
            a2 += wt * (float)vv[2]; a3 += wt * (float)vv[3];
        }
    }
    int hw = yi * CWB + xi;
    int tok = n * CZP + zi;
    v4bf rout;
    rout[0] = (bf16)a0; rout[1] = (bf16)a1; rout[2] = (bf16)a2; rout[3] = (bf16)a3;
    *(v4bf*)(mv + (((size_t)(b * CHW + hw) * CT + tok) * CC + c4)) = rout;
}

// ---------------------------------------------------------------------------
// 3) generic f32 -> bf16 convert
// ---------------------------------------------------------------------------
__global__ void k_cvt_bf16(const float* __restrict__ in, bf16* __restrict__ out, int n) {
    int i = blockIdx.x * blockDim.x + threadIdx.x;
    if (i < n) out[i] = (bf16)in[i];
}

// ---------------------------------------------------------------------------
// 4) WMMA bf16 GEMM:  D[M,N] = alpha * A[M,K] @ W[N,K]^T (+bias)(+resid)(gelu)
//    - 128x64 block tile, 8 wave32 waves, each wave 32x32 (2x2 WMMA 16x16x32)
//    - LDS double buffering via global_load_async_to_lds_b128 (ASYNCcnt)
//    flags: 1=bias  2=gelu  4=resid add  8=also write bf16 copy (pitch lddbf)
// ---------------------------------------------------------------------------
#if USE_ASYNC_LDS
#define ASYNC_WAIT_PREV() asm volatile("s_wait_asynccnt 3" ::: "memory")
#define ASYNC_WAIT_ALL()  asm volatile("s_wait_asynccnt 0" ::: "memory")
#else
#define ASYNC_WAIT_PREV()
#define ASYNC_WAIT_ALL()
#endif

__device__ __forceinline__ void gemm_load_stage(bf16* sA, bf16* sB,
                                                const bf16* A, const bf16* W,
                                                int lda, int ldw, int M, int N,
                                                int m0, int n0, int k0, int tid)
{
#if USE_ASYNC_LDS
    uint64_t abase = (uint64_t)(uintptr_t)A;
    uint64_t wbase = (uint64_t)(uintptr_t)W;
    uint32_t lA = (uint32_t)(uintptr_t)sA;
    uint32_t lB = (uint32_t)(uintptr_t)sB;
    #pragma unroll
    for (int c = 0; c < 2; c++) {                  // A: 128x32 bf16 = 512 x 16B
        int chunk = tid + c * 256;
        int r = chunk >> 2, kc = chunk & 3;
        int rg = m0 + r; rg = (rg < M) ? rg : (M - 1);
        uint32_t voff = (uint32_t)(((uint64_t)rg * (uint32_t)lda + (uint32_t)(k0 + kc * 8)) * 2ull);
        uint32_t la = lA + (uint32_t)((r * 32 + kc * 8) * 2);
        asm volatile("global_load_async_to_lds_b128 %0, %1, %2"
                     :: "v"(la), "v"(voff), "s"(abase) : "memory");
    }
    {                                              // W: 64x32 bf16 = 256 x 16B
        int chunk = tid;
        int r = chunk >> 2, kc = chunk & 3;
        int rg = n0 + r; rg = (rg < N) ? rg : (N - 1);
        uint32_t voff = (uint32_t)(((uint64_t)rg * (uint32_t)ldw + (uint32_t)(k0 + kc * 8)) * 2ull);
        uint32_t lb = lB + (uint32_t)((r * 32 + kc * 8) * 2);
        asm volatile("global_load_async_to_lds_b128 %0, %1, %2"
                     :: "v"(lb), "v"(voff), "s"(wbase) : "memory");
    }
#else
    #pragma unroll
    for (int c = 0; c < 2; c++) {
        int chunk = tid + c * 256;
        int r = chunk >> 2, kc = chunk & 3;
        int rg = m0 + r; rg = (rg < M) ? rg : (M - 1);
        *(uint4*)&sA[r * 32 + kc * 8] =
            *(const uint4*)&A[(size_t)rg * lda + k0 + kc * 8];
    }
    {
        int chunk = tid;
        int r = chunk >> 2, kc = chunk & 3;
        int rg = n0 + r; rg = (rg < N) ? rg : (N - 1);
        *(uint4*)&sB[r * 32 + kc * 8] =
            *(const uint4*)&W[(size_t)rg * ldw + k0 + kc * 8];
    }
#endif
}

__global__ __launch_bounds__(256)
void k_gemm_bf16(const bf16* __restrict__ A, int lda,
                 const bf16* __restrict__ W, int ldw,
                 const float* __restrict__ bias,
                 const float* __restrict__ resid,
                 float* __restrict__ D,
                 bf16* __restrict__ Dbf, int lddbf,
                 int M, int N, int K, float alpha, int flags)
{
    __shared__ bf16 sA[2][128 * 32];   // 16 KB
    __shared__ bf16 sB[2][64 * 32];    //  8 KB
    int tid  = threadIdx.x;
    int wid  = tid >> 5;
    int lane = tid & 31;
    int wm = wid >> 1;                 // 0..3 -> 32-row strip
    int wn = wid & 1;                  // 0..1 -> 32-col strip
    int lm = lane & 15;
    int hi = lane >> 4;
    int n0 = blockIdx.x * 64;
    int m0 = blockIdx.y * 128;

    v8f acc[2][2];
    const v8f vz = {0.f, 0.f, 0.f, 0.f, 0.f, 0.f, 0.f, 0.f};
    acc[0][0] = vz; acc[0][1] = vz; acc[1][0] = vz; acc[1][1] = vz;

    gemm_load_stage(sA[0], sB[0], A, W, lda, ldw, M, N, m0, n0, 0, tid);
    int buf = 0;
    for (int kt = 0; kt < K; kt += 32) {
        bool nxt = (kt + 32) < K;
        if (nxt) gemm_load_stage(sA[buf ^ 1], sB[buf ^ 1], A, W, lda, ldw, M, N, m0, n0, kt + 32, tid);
        if (nxt) { ASYNC_WAIT_PREV(); } else { ASYNC_WAIT_ALL(); }
        __syncthreads();

        const bf16* a_s = &sA[buf][0];
        const bf16* b_s = &sB[buf][0];
        v16bf af[2], bfr[2];
        #pragma unroll
        for (int i = 0; i < 2; i++) {
            // 16-bit A 16x32 layout: lanes 0-15 K chunk 0..7 / 16..23,
            // lanes 16-31 K chunk 8..15 / 24..31  (two 16B LDS reads)
            int row = wm * 32 + i * 16 + lm;
            uint4 u0 = *(const uint4*)&a_s[row * 32 + hi * 8];
            uint4 u1 = *(const uint4*)&a_s[row * 32 + 16 + hi * 8];
            uint4* dp = (uint4*)&af[i];
            dp[0] = u0; dp[1] = u1;
        }
        #pragma unroll
        for (int j = 0; j < 2; j++) {
            int row = wn * 32 + j * 16 + lm;
            uint4 u0 = *(const uint4*)&b_s[row * 32 + hi * 8];
            uint4 u1 = *(const uint4*)&b_s[row * 32 + 16 + hi * 8];
            uint4* dp = (uint4*)&bfr[j];
            dp[0] = u0; dp[1] = u1;
        }
        #pragma unroll
        for (int i = 0; i < 2; i++)
            #pragma unroll
            for (int j = 0; j < 2; j++)
                acc[i][j] = __builtin_amdgcn_wmma_f32_16x16x32_bf16(
                    false, af[i], false, bfr[j], (short)0, acc[i][j], false, false);
        __syncthreads();
        buf ^= 1;
    }

    // epilogue: C/D layout -> lane gives N=lm, VGPR r + 8*hi gives M
    #pragma unroll
    for (int i = 0; i < 2; i++) {
        #pragma unroll
        for (int j = 0; j < 2; j++) {
            #pragma unroll
            for (int r = 0; r < 8; r++) {
                int row = m0 + wm * 32 + i * 16 + hi * 8 + r;
                int col = n0 + wn * 32 + j * 16 + lm;
                if (row < M && col < N) {
                    float v = acc[i][j][r] * alpha;
                    if (flags & 1) v += bias[col];
                    if (flags & 4) v += resid[(size_t)row * N + col];
                    if (flags & 2) v = gelu_exact(v);
                    D[(size_t)row * N + col] = v;
                    if (flags & 8) Dbf[(size_t)row * lddbf + col] = (bf16)v;
                }
            }
        }
    }
}

// ---------------------------------------------------------------------------
// 5) GroupNorm(1,C) over whole sample; token-major input, bf16 output
// ---------------------------------------------------------------------------
__global__ void k_groupnorm1(const float* __restrict__ x,
                             const float* __restrict__ g,
                             const float* __restrict__ b,
                             bf16* __restrict__ out, int HW, int C)
{
    __shared__ float red[64];
    int bs = blockIdx.x;
    const float* xs = x + (size_t)bs * HW * C;
    int n = HW * C;
    float s = 0.f, s2 = 0.f;
    for (int i = threadIdx.x; i < n; i += blockDim.x) { float v = xs[i]; s += v; s2 += v * v; }
    s = wave_sum32(s); s2 = wave_sum32(s2);
    int w = threadIdx.x >> 5, l = threadIdx.x & 31, nw = blockDim.x >> 5;
    if (l == 0) { red[w] = s; red[32 + w] = s2; }
    __syncthreads();
    if (threadIdx.x == 0) {
        float a = 0.f, a2 = 0.f;
        for (int i = 0; i < nw; i++) { a += red[i]; a2 += red[32 + i]; }
        float m = a / (float)n;
        red[0] = m; red[1] = a2 / (float)n - m * m;
    }
    __syncthreads();
    float mean = red[0], inv = rsqrtf(red[1] + 1e-5f);
    bf16* os = out + (size_t)bs * HW * C;
    for (int i = threadIdx.x; i < n; i += blockDim.x) {
        int c = i & (C - 1);
        os[i] = (bf16)((xs[i] - mean) * inv * g[c] + b[c]);
    }
}

// ---------------------------------------------------------------------------
// 6) LayerNorm over C=128 per token; bf16 output (wave per token)
// ---------------------------------------------------------------------------
__global__ void k_layernorm_bf16(const float* __restrict__ x,
                                 const float* __restrict__ g,
                                 const float* __restrict__ b,
                                 bf16* __restrict__ out, int ntok)
{
    int tok = blockIdx.x * 8 + (threadIdx.x >> 5);
    if (tok >= ntok) return;
    int lane = threadIdx.x & 31;
    const float* xs = x + (size_t)tok * CC;
    float v[4]; float s = 0.f;
    #pragma unroll
    for (int i = 0; i < 4; i++) { v[i] = xs[lane + 32 * i]; s += v[i]; }
    s = wave_sum32(s) * (1.0f / CC);
    float s2 = 0.f;
    #pragma unroll
    for (int i = 0; i < 4; i++) { float d = v[i] - s; s2 += d * d; }
    s2 = wave_sum32(s2) * (1.0f / CC);
    float inv = rsqrtf(s2 + 1e-6f);
    #pragma unroll
    for (int i = 0; i < 4; i++) {
        int c = lane + 32 * i;
        out[(size_t)tok * CC + c] = (bf16)((v[i] - s) * inv * g[c] + b[c]);
    }
}

// ---------------------------------------------------------------------------
// 7) row softmax (n<=128); optional zero-padded bf16 copy (pitch ldbf)
// ---------------------------------------------------------------------------
__global__ void k_softmax(const float* __restrict__ x, float* __restrict__ y,
                          bf16* __restrict__ ybf, int rows, int n, int ld, int ldbf)
{
    int row = blockIdx.x * 8 + (threadIdx.x >> 5);
    if (row >= rows) return;
    int lane = threadIdx.x & 31;
    float v[4]; float mx = -3.4e38f;
    #pragma unroll
    for (int i = 0; i < 4; i++) {
        int c = lane + 32 * i;
        v[i] = (c < n) ? x[(size_t)row * ld + c] : -3.4e38f;
        mx = fmaxf(mx, v[i]);
    }
    mx = wave_max32(mx);
    float s = 0.f;
    #pragma unroll
    for (int i = 0; i < 4; i++) { v[i] = (v[i] > -1e37f) ? __expf(v[i] - mx) : 0.f; s += v[i]; }
    s = wave_sum32(s);
    float inv = 1.0f / s;
    #pragma unroll
    for (int i = 0; i < 4; i++) {
        int c = lane + 32 * i;
        float o = v[i] * inv;
        if (c < n) y[(size_t)row * ld + c] = o;
        if (ybf && c < ldbf) ybf[(size_t)row * ldbf + c] = (bf16)((c < n) ? o : 0.f);
    }
}

// ---------------------------------------------------------------------------
// 8) attn-apply over mv tokens: out[tok,c] = sum_t attn[tok,t]*mv[tok,t,c]
//    (mv is L2-resident on MI455X: 79MB << 192MB)
// ---------------------------------------------------------------------------
__global__ __launch_bounds__(256)
void k_attn_apply(const float* __restrict__ attn, const bf16* __restrict__ mv,
                  float* __restrict__ out)
{
    int tok = blockIdx.x * 8 + (threadIdx.x >> 5);
    if (tok >= CBHW) return;
    int lane = threadIdx.x & 31;
    int c4 = lane * 4;
    const float* ar = attn + (size_t)tok * CT;
    const bf16* mr = mv + (size_t)tok * CT * CC + c4;
    float a0 = 0.f, a1 = 0.f, a2 = 0.f, a3 = 0.f;
    for (int t = 0; t < CT; t++) {
        float w = ar[t];
        v4bf v = *(const v4bf*)(mr + (size_t)t * CC);
        a0 += w * (float)v[0]; a1 += w * (float)v[1];
        a2 += w * (float)v[2]; a3 += w * (float)v[3];
    }
    float* o = out + (size_t)tok * CC + c4;
    o[0] = a0; o[1] = a1; o[2] = a2; o[3] = a3;
}

// ---------------------------------------------------------------------------
// 9) im2col for 4x4 stride-4 SR conv (token-major bf16 in/out)
// ---------------------------------------------------------------------------
__global__ void k_im2col4x4(const bf16* __restrict__ h, bf16* __restrict__ out)
{
    int idx = blockIdx.x * blockDim.x + threadIdx.x;   // 200*2048
    if (idx >= CBSR * 2048) return;
    int row = idx >> 11, e = idx & 2047;
    int b = row / CSR, pr = row - b * CSR;
    int py = pr / 10, px = pr - py * 10;
    int ci = e >> 4, kh = (e >> 2) & 3, kw = e & 3;
    int y = py * 4 + kh, x = px * 4 + kw;
    out[idx] = h[((size_t)(b * CHW + y * CWB + x)) * CC + ci];
}

// ---------------------------------------------------------------------------
// 10) v (B*100,128) bf16 -> vT (B,128,128) bf16 with zero pad rows 100..127
// ---------------------------------------------------------------------------
__global__ void k_transpose_pad(const bf16* __restrict__ v, bf16* __restrict__ vt)
{
    int idx = blockIdx.x * blockDim.x + threadIdx.x;   // B*128*128
    if (idx >= CB * 128 * 128) return;
    int b = idx >> 14, r = idx & 16383;
    int c = r >> 7, m = r & 127;
    vt[idx] = (m < CSR) ? v[(size_t)(b * CSR + m) * 128 + c] : (bf16)0.f;
}

// ---------------------------------------------------------------------------
// 11) depthwise 3x3 SAME on (B,512,40,40) token-major f32 -> gelu -> bf16
// ---------------------------------------------------------------------------
__global__ void k_dwconv3x3_gelu(const float* __restrict__ f,
                                 const float* __restrict__ w,
                                 const float* __restrict__ bias,
                                 bf16* __restrict__ out)
{
    int idx = blockIdx.x * blockDim.x + threadIdx.x;   // B*HW*128 chunks of 4ch
    if (idx >= CB * CHW * (CH4 / 4)) return;
    int c4 = (idx & 127) * 4;
    int t = idx >> 7;
    int b = t / CHW, hw = t - b * CHW;
    int y = hw / CWB, x = hw - y * CWB;
    float acc0 = bias[c4], acc1 = bias[c4 + 1], acc2 = bias[c4 + 2], acc3 = bias[c4 + 3];
    #pragma unroll
    for (int dy = -1; dy <= 1; dy++) {
        #pragma unroll
        for (int dx = -1; dx <= 1; dx++) {
            int yy = y + dy, xx = x + dx;
            if (yy < 0 || yy >= CHB || xx < 0 || xx >= CWB) continue;
            const float* fp = f + ((size_t)(b * CHW + yy * CWB + xx)) * CH4 + c4;
            int wi = (dy + 1) * 3 + (dx + 1);
            acc0 += fp[0] * w[(c4 + 0) * 9 + wi];
            acc1 += fp[1] * w[(c4 + 1) * 9 + wi];
            acc2 += fp[2] * w[(c4 + 2) * 9 + wi];
            acc3 += fp[3] * w[(c4 + 3) * 9 + wi];
        }
    }
    bf16* op = out + (size_t)t * CH4 + c4;
    op[0] = (bf16)gelu_exact(acc0); op[1] = (bf16)gelu_exact(acc1);
    op[2] = (bf16)gelu_exact(acc2); op[3] = (bf16)gelu_exact(acc3);
}

// ---------------------------------------------------------------------------
// 12) misc elementwise
// ---------------------------------------------------------------------------
__global__ void k_add_inplace(float* __restrict__ y, const float* __restrict__ x, int n) {
    int i = blockIdx.x * blockDim.x + threadIdx.x;
    if (i < n) y[i] += x[i];
}
__global__ void k_init_query(const float* __restrict__ bev, float* __restrict__ q) {
    int idx = blockIdx.x * blockDim.x + threadIdx.x;   // B*HW*C token-major
    if (idx >= CBHW * CC) return;
    int c = idx & (CC - 1);
    int hw = (idx >> 7) % CHW;
    q[idx] = bev[c * CHW + hw];
}

// ---------------------------------------------------------------------------
// 13) fused bilinear 8x resize + out_proj(128->8) + circle mask
// ---------------------------------------------------------------------------
__global__ void k_resize_out(const float* __restrict__ q,   // (B,HW,C) tok f32
                             const float* __restrict__ w,   // (8,128)
                             const float* __restrict__ bias,// (8,)
                             float* __restrict__ out)       // (B,8,320,320)
{
    __shared__ float sw[8 * 128];
    __shared__ float sb[8];
    for (int i = threadIdx.x; i < 8 * 128; i += blockDim.x) sw[i] = w[i];
    if (threadIdx.x < 8) sb[threadIdx.x] = bias[threadIdx.x];
    __syncthreads();
    int idx = blockIdx.x * blockDim.x + threadIdx.x;
    if (idx >= CB * 320 * 320) return;
    int b = idx / (320 * 320);
    int pix = idx - b * 320 * 320;
    int y = pix / 320, x = pix - y * 320;
    float sy = ((float)y + 0.5f) * 0.125f - 0.5f;
    float sx = ((float)x + 0.5f) * 0.125f - 0.5f;
    int y0 = (int)floorf(sy), x0 = (int)floorf(sx);
    float fy = sy - (float)y0, fx = sx - (float)x0;
    int y0c = min(max(y0, 0), CHB - 1), y1c = min(max(y0 + 1, 0), CHB - 1);
    int x0c = min(max(x0, 0), CWB - 1), x1c = min(max(x0 + 1, 0), CWB - 1);
    const float* qb = q + (size_t)b * CHW * CC;
    float acc[8];
    #pragma unroll
    for (int o = 0; o < 8; o++) acc[o] = sb[o];
    for (int c = 0; c < CC; c++) {
        float v00 = qb[(size_t)(y0c * CWB + x0c) * CC + c];
        float v01 = qb[(size_t)(y0c * CWB + x1c) * CC + c];
        float v10 = qb[(size_t)(y1c * CWB + x0c) * CC + c];
        float v11 = qb[(size_t)(y1c * CWB + x1c) * CC + c];
        float v = v00 * (1.f - fy) * (1.f - fx) + v01 * (1.f - fy) * fx
                + v10 * fy * (1.f - fx)         + v11 * fy * fx;
        #pragma unroll
        for (int o = 0; o < 8; o++) acc[o] += v * sw[o * 128 + c];
    }
    float dy = (float)y - 159.5f, dx = (float)x - 159.5f;
    float mask = (dy * dy + dx * dx <= 160.f * 160.f) ? 1.f : 0.f;
    #pragma unroll
    for (int o = 0; o < 8; o++)
        out[((size_t)(b * 8 + o) * 320 + y) * 320 + x] = acc[o] * mask;
}

// ---------------------------------------------------------------------------
// Host orchestration
// ---------------------------------------------------------------------------
static inline unsigned cdiv(unsigned a, unsigned b) { return (a + b - 1) / b; }

extern "C" void kernel_launch(void* const* d_in, const int* in_sizes, int n_in,
                              void* d_out, int out_size, void* d_ws, size_t ws_size,
                              hipStream_t stream)
{
    (void)in_sizes; (void)out_size; (void)ws_size;
    auto din = [&](int i) -> const float* {
        return (const float*)d_in[(i < n_in) ? i : 0];
    };
    const float* fmaps = din(0);
    const float* extr  = din(1);
    const float* intr  = din(2);
    const float* res   = din(3);
    const float* mpp   = din(4);
    const float* bev   = din(5);
    // per-block param leaf offsets (jax sorted-dict flatten):
    // 0 gn1.b 1 gn1.g 2 gn2.b 3 gn2.g 4 mlp1.b 5 mlp1.w 6 mlp2.b 7 mlp2.w
    // 8 sa.dw_b 9 sa.dw_w 10 sa.fc1.b 11 sa.fc1.w 12 sa.fc2.b 13 sa.fc2.w
    // 14 sa.k.b 15 sa.k.w 16 sa.ln1.b 17 sa.ln1.g 18 sa.ln2.b 19 sa.ln2.g
    // 20 sa.o.b 21 sa.o.w 22 sa.q.b 23 sa.q.w 24 sa.sr_b 25 sa.sr_ln.b
    // 26 sa.sr_ln.g 27 sa.sr_w 28 sa.v.b 29 sa.v.w 30 tl.b 31 tl.w
    auto blk = [&](int bi, int off) -> const float* { return din(6 + 32 * bi + off); };
    const float* opb = din(102);
    const float* opw = din(103);

    // ---- workspace carve ----
    char* wsB = (char*)d_ws;
    size_t off = 0;
    auto alloc = [&](size_t bytes) -> void* {
        void* p = wsB + off;
        off = (off + bytes + 255) & ~(size_t)255;
        return p;
    };
    bf16*  fm_cl   = (bf16*)alloc((size_t)CB * CN * CHF * CWF * CC * 2);     // 138 MB
    bf16*  mv      = (bf16*)alloc((size_t)CBHW * CT * CC * 2);               //  79 MB
    float* qA      = (float*)alloc((size_t)CBHW * CC * 4);
    float* qB      = (float*)alloc((size_t)CBHW * CC * 4);
    float* logitsA = (float*)alloc((size_t)CBHW * CT * 4);
    float* logitsB = (float*)alloc((size_t)CBHW * CT * 4);
    float* attn    = (float*)alloc((size_t)CBHW * CT * 4);
    bf16*  gn_bf   = (bf16*)alloc((size_t)CBHW * CC * 2);
    float* xtok    = (float*)alloc((size_t)CBHW * CC * 4);
    float* scrD    = (float*)alloc((size_t)CBHW * CH4 * 4);                  // 6.6 MB
    bf16*  h_bf    = (bf16*)alloc((size_t)CBHW * CC * 2);
    bf16*  q_bf    = (bf16*)alloc((size_t)CBHW * CC * 2);
    bf16*  k_bf    = (bf16*)alloc((size_t)CBSR * 128 * 2);
    bf16*  v_bf    = (bf16*)alloc((size_t)CBSR * 128 * 2);
    bf16*  vT_bf   = (bf16*)alloc((size_t)CB * 128 * 128 * 2);
    bf16*  patches = (bf16*)alloc((size_t)CBSR * 2048 * 2);
    float* hrpre   = (float*)alloc((size_t)CBSR * CC * 4);
    bf16*  hr_bf   = (bf16*)alloc((size_t)CBSR * CC * 2);
    float* scores  = (float*)alloc((size_t)CBHW * CSR * 4);
    float* attn2   = (float*)alloc((size_t)CBHW * CSR * 4);
    bf16*  attn2bf = (bf16*)alloc((size_t)CBHW * 128 * 2);
    float* ctxf    = (float*)alloc((size_t)CBHW * CC * 4);
    bf16*  ctx_bf  = (bf16*)alloc((size_t)CBHW * CC * 2);
    float* h2      = (float*)alloc((size_t)CBHW * CC * 4);
    bf16*  f_bf    = (bf16*)alloc((size_t)CBHW * CC * 2);
    float* f1      = (float*)alloc((size_t)CBHW * CH4 * 4);
    bf16*  f2_bf   = (bf16*)alloc((size_t)CBHW * CH4 * 2);
    float* segout  = (float*)alloc((size_t)CBHW * CC * 4);
    bf16*  m1_bf   = (bf16*)alloc((size_t)CBHW * CH4 * 2);
    bf16 *w_tl[3], *w_q[3], *w_k[3], *w_v[3], *w_o[3], *w_sr[3], *w_f1[3], *w_f2[3], *w_m1[3], *w_m2[3];
    for (int bi = 0; bi < 3; bi++) {
        w_tl[bi] = (bf16*)alloc((size_t)CT * CC * 2);
        w_q[bi]  = (bf16*)alloc((size_t)CC * CC * 2);
        w_k[bi]  = (bf16*)alloc((size_t)CC * CC * 2);
        w_v[bi]  = (bf16*)alloc((size_t)CC * CC * 2);
        w_o[bi]  = (bf16*)alloc((size_t)CC * CC * 2);
        w_sr[bi] = (bf16*)alloc((size_t)CC * 2048 * 2);
        w_f1[bi] = (bf16*)alloc((size_t)CH4 * CC * 2);
        w_f2[bi] = (bf16*)alloc((size_t)CC * CH4 * 2);
        w_m1[bi] = (bf16*)alloc((size_t)CH4 * CC * 2);
        w_m2[bi] = (bf16*)alloc((size_t)CC * CH4 * 2);
    }

    auto cvt = [&](const float* src, bf16* dst, int n) {
        k_cvt_bf16<<<cdiv(n, 256), 256, 0, stream>>>(src, dst, n);
    };
    auto gemm = [&](const bf16* A, int lda, const bf16* W, int ldw,
                    const float* bias, const float* resid,
                    float* D, bf16* Dbf, int lddbf,
                    int M, int N, int K, float alpha, int flags) {
        dim3 g(cdiv(N, 64), cdiv(M, 128));
        k_gemm_bf16<<<g, 256, 0, stream>>>(A, lda, W, ldw, bias, resid, D, Dbf, lddbf,
                                           M, N, K, alpha, flags);
    };

    // ---- weight conversion (small, once per launch) ----
    for (int bi = 0; bi < 3; bi++) {
        cvt(blk(bi, 31), w_tl[bi], CT * CC);
        cvt(blk(bi, 23), w_q[bi],  CC * CC);
        cvt(blk(bi, 15), w_k[bi],  CC * CC);
        cvt(blk(bi, 29), w_v[bi],  CC * CC);
        cvt(blk(bi, 21), w_o[bi],  CC * CC);
        cvt(blk(bi, 27), w_sr[bi], CC * 2048);
        cvt(blk(bi, 11), w_f1[bi], CH4 * CC);
        cvt(blk(bi, 13), w_f2[bi], CC * CH4);
        cvt(blk(bi, 5),  w_m1[bi], CH4 * CC);
        cvt(blk(bi, 7),  w_m2[bi], CC * CH4);
    }

    // ---- stage 1: layout change + mv-token gather ----
    {
        dim3 g(cdiv(CHF * CWF, 32), CC / 32, CB * CN);
        k_nchw_to_nhwc_bf16<<<g, 256, 0, stream>>>(fmaps, fm_cl, CC, CHF * CWF);
    }
    {
        int waves = CB * CN * CHB * CWB * CZP;
        k_mv_gather<<<cdiv(waves, 8), 256, 0, stream>>>(fm_cl, extr, intr, res, mpp, mv);
    }
    k_init_query<<<cdiv(CBHW * CC, 256), 256, 0, stream>>>(bev, qA);

    // ---- stage 2: decoder blocks ----
    float* logitsP = logitsA;
    float* logitsC = logitsB;
    const float inv_sqrt_c = 0.08838834764831845f;   // 1/sqrt(128)
    for (int bi = 0; bi < 3; bi++) {
        // cross-attention into mv tokens
        k_groupnorm1<<<CB, 1024, 0, stream>>>(qA, blk(bi, 1), blk(bi, 0), gn_bf, CHW, CC);
        gemm(gn_bf, CC, w_tl[bi], CC, blk(bi, 30), (bi ? logitsP : nullptr),
             logitsC, nullptr, 0, CBHW, CT, CC, 1.f, 1 | (bi ? 4 : 0));
        k_softmax<<<cdiv(CBHW, 8), 256, 0, stream>>>(logitsC, attn, nullptr, CBHW, CT, CT, 0);
        k_attn_apply<<<cdiv(CBHW, 8), 256, 0, stream>>>(attn, mv, xtok);

        // segformer layer
        k_layernorm_bf16<<<cdiv(CBHW, 8), 256, 0, stream>>>(xtok, blk(bi, 17), blk(bi, 16), h_bf, CBHW);
        gemm(h_bf, CC, w_q[bi], CC, blk(bi, 22), nullptr, scrD, q_bf, CC, CBHW, CC, CC, 1.f, 1 | 8);
        k_im2col4x4<<<cdiv(CBSR * 2048, 256), 256, 0, stream>>>(h_bf, patches);
        gemm(patches, 2048, w_sr[bi], 2048, blk(bi, 24), nullptr, hrpre, nullptr, 0,
             CBSR, CC, 2048, 1.f, 1);
        k_layernorm_bf16<<<cdiv(CBSR, 8), 256, 0, stream>>>(hrpre, blk(bi, 26), blk(bi, 25), hr_bf, CBSR);
        gemm(hr_bf, CC, w_k[bi], CC, blk(bi, 14), nullptr, scrD, k_bf, 128, CBSR, CC, CC, 1.f, 1 | 8);
        gemm(hr_bf, CC, w_v[bi], CC, blk(bi, 28), nullptr, scrD, v_bf, 128, CBSR, CC, CC, 1.f, 1 | 8);
        k_transpose_pad<<<cdiv(CB * 128 * 128, 256), 256, 0, stream>>>(v_bf, vT_bf);
        for (int b = 0; b < CB; b++) {
            gemm(q_bf + (size_t)b * CHW * CC, CC, k_bf + (size_t)b * CSR * 128, 128,
                 nullptr, nullptr, scores + (size_t)b * CHW * CSR, nullptr, 0,
                 CHW, CSR, CC, inv_sqrt_c, 0);
        }
        k_softmax<<<cdiv(CBHW, 8), 256, 0, stream>>>(scores, attn2, attn2bf, CBHW, CSR, CSR, 128);
        for (int b = 0; b < CB; b++) {
            gemm(attn2bf + (size_t)b * CHW * 128, 128, vT_bf + (size_t)b * 128 * 128, 128,
                 nullptr, nullptr, ctxf + (size_t)b * CHW * CC,
                 ctx_bf + (size_t)b * CHW * CC, CC, CHW, CC, 128, 1.f, 8);
        }
        gemm(ctx_bf, CC, w_o[bi], CC, blk(bi, 20), xtok, h2, nullptr, 0, CBHW, CC, CC, 1.f, 1 | 4);
        k_layernorm_bf16<<<cdiv(CBHW, 8), 256, 0, stream>>>(h2, blk(bi, 19), blk(bi, 18), f_bf, CBHW);
        gemm(f_bf, CC, w_f1[bi], CC, blk(bi, 10), nullptr, f1, nullptr, 0, CBHW, CH4, CC, 1.f, 1);
        k_dwconv3x3_gelu<<<cdiv(CB * CHW * (CH4 / 4), 256), 256, 0, stream>>>(
            f1, blk(bi, 9), blk(bi, 8), f2_bf);
        gemm(f2_bf, CH4, w_f2[bi], CH4, blk(bi, 12), h2, segout, nullptr, 0,
             CBHW, CC, CH4, 1.f, 1 | 4);
        k_add_inplace<<<cdiv(CBHW * CC, 256), 256, 0, stream>>>(segout, qA, CBHW * CC);

        // channel MLP
        k_groupnorm1<<<CB, 1024, 0, stream>>>(segout, blk(bi, 3), blk(bi, 2), gn_bf, CHW, CC);
        gemm(gn_bf, CC, w_m1[bi], CC, blk(bi, 4), nullptr, scrD, m1_bf, CH4,
             CBHW, CH4, CC, 1.f, 1 | 2 | 8);
        gemm(m1_bf, CH4, w_m2[bi], CH4, blk(bi, 6), segout, qB, nullptr, 0,
             CBHW, CC, CH4, 1.f, 1 | 4);

        float* t = qA; qA = qB; qB = t;
        t = logitsP; logitsP = logitsC; logitsC = t;
    }

    // ---- stage 3: upsample + projection + circle mask ----
    k_resize_out<<<cdiv(CB * 320 * 320, 256), 256, 0, stream>>>(qA, opw, opb, (float*)d_out);
}
